// GGRUCell_27367531610436
// MI455X (gfx1250) — matrix operand order repaired and checked by reading, not compiled, not run on hardware
//
#include <hip/hip_runtime.h>

// ---------------------------------------------------------------------------
// GGRU cell, MI455X (gfx1250, wave32, WMMA).
//
// Roofline: dominant op is h@u_W = 204800x256x256 GEMM (26.8 GFLOP) over a
// 210MB h tensor -> 128 FLOP/byte.  HBM floor ~9us @ 23.3TB/s; bf16 WMMA
// compute ~5us -> memory-bound with v_wmma_f32_16x16x32_bf16: convert
// f32->bf16 on the fly (packed cvt + b128 LDS stores) and stream h once
// through a double-buffered LDS tile.
// Kernel 0: transpose+convert weights to bf16 scratch (1MB d_ws) so WMMA
//           B-fragments are contiguous 32B-per-lane global loads.
// Kernel 1: fused  sigmoid(g@w_W + h@u_W + u_b)*mask -> softmax(N) -> h_avg.
//           Grid 64x2 (16-batch x 128-col tiles), 256 thr = 8 waves,
//           u_W^T fragments pinned in 64 VGPRs, dual WMMA accumulators.
// Kernel 2: GRU tail: zr=sigmoid(g@W_W + h_avg@U_W + U_b),
//           gt=tanh([r*g, h_avg]@V_W + V_b), out=(1-z)g + z*gt.  h_avg is
//           staged in d_out and fully consumed into LDS before overwrite.
// ---------------------------------------------------------------------------

#define BDIM 1024
#define NW   200
#define HD   256

typedef __bf16 bf16_t;
typedef bf16_t v16bf __attribute__((ext_vector_type(16)));
typedef bf16_t v8bf  __attribute__((ext_vector_type(8)));
typedef float  v8f   __attribute__((ext_vector_type(8)));

// bf16 scratch layout (element offsets): transposed weights, [outCol][K]
#define OFF_UT   0                              // u_W^T  [256][256]
#define OFF_WT   (256 * 256)                    // w_W^T  [256][256]
#define OFF_WWT  (2 * 256 * 256)                // W_W^T  [512][256]
#define OFF_UWT  (2 * 256 * 256 + 512 * 256)    // U_W^T  [512][256]
#define OFF_VWT  (2 * 256 * 256 + 2 * 512 * 256)// V_W^T  [256][512]

__device__ __forceinline__ v16bf make_v16(v8bf lo, v8bf hi) {
  return __builtin_shufflevector(lo, hi, 0, 1, 2, 3, 4, 5, 6, 7,
                                 8, 9, 10, 11, 12, 13, 14, 15);
}

// 8x f32 -> packed bf16x8 (v_cvt_pk_bf16_f32 path), for 16B LDS stores
__device__ __forceinline__ v8bf cvt_v8(float4 a, float4 b) {
  v8f f = {a.x, a.y, a.z, a.w, b.x, b.y, b.z, b.w};
  return __builtin_convertvector(f, v8bf);
}

// B fragment (32x16 bf16): lane holds col n, K = kbase + half*16 .. +15
// -> 32 contiguous bytes from the pre-transposed weight (two b128 loads).
__device__ __forceinline__ v16bf load_b_frag(const bf16_t* __restrict__ base,
                                             int n, int kdim, int kbase,
                                             int half) {
  const bf16_t* p = base + (size_t)n * kdim + kbase + half * 16;
  v8bf lo = *(const v8bf*)p;
  v8bf hi = *(const v8bf*)(p + 8);
  return make_v16(lo, hi);
}

// A fragment (16x32 bf16) from row-major LDS tile: lane row m,
// runs K = kbase+half*16..+15 (2x ds_load_b128).
__device__ __forceinline__ v16bf load_a_frag(const bf16_t* row, int kbase,
                                             int half) {
  const bf16_t* p = row + kbase + half * 16;
  v8bf lo = *(const v8bf*)p;
  v8bf hi = *(const v8bf*)(p + 8);
  return make_v16(lo, hi);
}

__device__ __forceinline__ v8f wmma_bf16(v16bf a, v16bf b, v8f c) {
  return __builtin_amdgcn_wmma_f32_16x16x32_bf16(false, a, false, b,
                                                 (short)0, c, false, false);
}

// ---------------------------------------------------------------------------
// Kernel 0: f32 -> bf16, transposed weight copies in d_ws
// ---------------------------------------------------------------------------
__global__ __launch_bounds__(256) void convert_weights(
    const float* __restrict__ w_W, const float* __restrict__ u_W,
    const float* __restrict__ W_W, const float* __restrict__ U_W,
    const float* __restrict__ V_W, bf16_t* __restrict__ ws) {
  const int tid = blockIdx.x * blockDim.x + threadIdx.x;
  const int stride = gridDim.x * blockDim.x;
  for (int i = tid; i < HD * HD; i += stride) {        // u_W^T, w_W^T
    int n = i >> 8, k = i & 255;
    ws[OFF_UT + i] = (bf16_t)u_W[(size_t)k * HD + n];
    ws[OFF_WT + i] = (bf16_t)w_W[(size_t)k * HD + n];
  }
  for (int i = tid; i < 2 * HD * HD; i += stride) {    // W_W^T, U_W^T [512][256]
    int n = i >> 8, k = i & 255;
    ws[OFF_WWT + i] = (bf16_t)W_W[(size_t)k * (2 * HD) + n];
    ws[OFF_UWT + i] = (bf16_t)U_W[(size_t)k * (2 * HD) + n];
  }
  for (int i = tid; i < 2 * HD * HD; i += stride) {    // V_W^T [256][512]
    int n = i >> 9, k = i & 511;
    ws[OFF_VWT + i] = (bf16_t)V_W[(size_t)k * HD + n];
  }
}

// ---------------------------------------------------------------------------
// Kernel 1: fused attention pool -> h_avg (written to d_out)
// ---------------------------------------------------------------------------
__global__ __launch_bounds__(256) void attn_pool_kernel(
    const float* __restrict__ g, const float* __restrict__ h,
    const int* __restrict__ mask, const float* __restrict__ u_b,
    const bf16_t* __restrict__ ws, float* __restrict__ havg) {
  __shared__ __align__(16) bf16_t hs[2][16][HD];   // double-buffered h tile
  __shared__ __align__(16) bf16_t gs[16][HD];      // g tile (bf16)
  __shared__ __align__(16) float ms[NW * 16];      // mask as float, [n][row]

  const int tid   = threadIdx.x;
  const int wave  = tid >> 5;
  const int lane  = tid & 31;
  const int half  = lane >> 4;     // WMMA half-wave
  const int m     = lane & 15;     // row (A) / col (B,C) within fragment
  const int bbase = blockIdx.x * 16;
  const int n0    = blockIdx.y * 128 + wave * 16;  // this wave's output cols
  const int colg  = n0 + m;

  // cooperative tile mapping: thread -> (row, 16 consecutive k)
  const int prow = tid >> 4;
  const int pk   = (tid & 15) << 4;

  for (int i = tid; i < NW * 16; i += 256) {
    int n = i >> 4, r = i & 15;
    ms[i] = (float)mask[(size_t)(bbase + r) * NW + n];
  }
  {
    const float* gp = g + (size_t)(bbase + prow) * HD + pk;
    float4 a0 = *(const float4*)(gp + 0);
    float4 a1 = *(const float4*)(gp + 4);
    float4 a2 = *(const float4*)(gp + 8);
    float4 a3 = *(const float4*)(gp + 12);
    *(v8bf*)&gs[prow][pk]     = cvt_v8(a0, a1);
    *(v8bf*)&gs[prow][pk + 8] = cvt_v8(a2, a3);
  }
  __syncthreads();

  // a = g @ w_W (+ u_b), C-layout registers, computed once per wave
  v8f acc = {};
#pragma unroll
  for (int kc = 0; kc < 8; ++kc) {
    acc = wmma_bf16(load_a_frag(&gs[m][0], kc * 32, half),
                    load_b_frag(ws + OFF_WT, colg, HD, kc * 32, half), acc);
  }
  const float ub = u_b[colg];
  float aR[8];
#pragma unroll
  for (int i = 0; i < 8; ++i) aR[i] = acc[i] + ub;

  // u_W^T B fragments: pinned in 64 VGPRs for the entire N loop
  v16bf uF[8];
#pragma unroll
  for (int kc = 0; kc < 8; ++kc)
    uF[kc] = load_b_frag(ws + OFF_UT, colg, HD, kc * 32, half);

  // prefetch word 0
  const float* hp = h + ((size_t)(bbase + prow) * NW + 0) * HD + pk;
  float4 p0 = *(const float4*)(hp + 0);
  float4 p1 = *(const float4*)(hp + 4);
  float4 p2 = *(const float4*)(hp + 8);
  float4 p3 = *(const float4*)(hp + 12);
  *(v8bf*)&hs[0][prow][pk]     = cvt_v8(p0, p1);
  *(v8bf*)&hs[0][prow][pk + 8] = cvt_v8(p2, p3);
  __syncthreads();

  float num[8] = {0.f, 0.f, 0.f, 0.f, 0.f, 0.f, 0.f, 0.f};
  float den[8] = {0.f, 0.f, 0.f, 0.f, 0.f, 0.f, 0.f, 0.f};

  for (int n = 0; n < NW; ++n) {
    const int cur = n & 1;
    if (n + 1 < NW) {  // issue next word's global loads before compute
      const float* hq = h + ((size_t)(bbase + prow) * NW + (n + 1)) * HD + pk;
      p0 = *(const float4*)(hq + 0);
      p1 = *(const float4*)(hq + 4);
      p2 = *(const float4*)(hq + 8);
      p3 = *(const float4*)(hq + 12);
    }

    // dual accumulators: halve the WMMA C-chain dependency depth
    v8f ca = {}, cb = {};
#pragma unroll
    for (int kc = 0; kc < 8; kc += 2) {
      ca = wmma_bf16(load_a_frag(&hs[cur][m][0], kc * 32, half), uF[kc], ca);
      cb = wmma_bf16(load_a_frag(&hs[cur][m][0], (kc + 1) * 32, half),
                     uF[kc + 1], cb);
    }

    // per-half-wave mask rows as one 32B vector load
    v8f mrow = *(const v8f*)&ms[n * 16 + half * 8];

    // f = exp(sigmoid(s + a) * mask); masked entries contribute exp(0)=1,
    // exactly matching softmax-after-mask in the reference.
#pragma unroll
    for (int i = 0; i < 8; ++i) {
      const int row = half * 8 + i;
      const float sg =
          1.0f / (1.0f + __expf(-(ca[i] + cb[i] + aR[i])));
      const float e = __expf(sg * mrow[i]);
      den[i] += e;
      num[i] += (float)hs[cur][row][colg] * e;   // weighted sum reuses tile
    }

    if (n + 1 < NW) {  // fill the other buffer (prior reads barriered)
      *(v8bf*)&hs[cur ^ 1][prow][pk]     = cvt_v8(p0, p1);
      *(v8bf*)&hs[cur ^ 1][prow][pk + 8] = cvt_v8(p2, p3);
    }
    __syncthreads();
  }

#pragma unroll
  for (int i = 0; i < 8; ++i) {
    const int row = half * 8 + i;
    havg[(size_t)(bbase + row) * HD + colg] = num[i] / den[i];
  }
}

// ---------------------------------------------------------------------------
// Kernel 2: GRU tail.  `out` holds h_avg on entry, final result on exit.
// ---------------------------------------------------------------------------
__global__ __launch_bounds__(256) void gru_gate_kernel(
    const float* __restrict__ g, const float* __restrict__ U_b,
    const float* __restrict__ V_b, const bf16_t* __restrict__ ws,
    float* __restrict__ out) {
  __shared__ __align__(16) bf16_t gA[16][HD];
  __shared__ __align__(16) bf16_t hA[16][HD];
  __shared__ __align__(16) bf16_t rgA[16][HD];
  __shared__ __align__(16) float zrS[16][2 * HD];

  const int tid   = threadIdx.x;
  const int wave  = tid >> 5;
  const int lane  = tid & 31;
  const int half  = lane >> 4;
  const int m     = lane & 15;
  const int bbase = blockIdx.x * 16;
  const int prow  = tid >> 4;
  const int pk    = (tid & 15) << 4;

  {
    const float* gp = g + (size_t)(bbase + prow) * HD + pk;
    const float* ap = out + (size_t)(bbase + prow) * HD + pk;  // h_avg
    float4 a0 = *(const float4*)(gp + 0);
    float4 a1 = *(const float4*)(gp + 4);
    float4 a2 = *(const float4*)(gp + 8);
    float4 a3 = *(const float4*)(gp + 12);
    float4 b0 = *(const float4*)(ap + 0);
    float4 b1 = *(const float4*)(ap + 4);
    float4 b2 = *(const float4*)(ap + 8);
    float4 b3 = *(const float4*)(ap + 12);
    *(v8bf*)&gA[prow][pk]     = cvt_v8(a0, a1);
    *(v8bf*)&gA[prow][pk + 8] = cvt_v8(a2, a3);
    *(v8bf*)&hA[prow][pk]     = cvt_v8(b0, b1);
    *(v8bf*)&hA[prow][pk + 8] = cvt_v8(b2, b3);
  }
  __syncthreads();

  // zr = sigmoid(g@W_W + h_avg@U_W + U_b); wave covers 64 of 512 cols
  for (int t = 0; t < 4; ++t) {
    const int col = wave * 64 + t * 16 + m;
    v8f ca = {}, cb = {};
#pragma unroll
    for (int kc = 0; kc < 8; ++kc)
      ca = wmma_bf16(load_a_frag(&gA[m][0], kc * 32, half),
                     load_b_frag(ws + OFF_WWT, col, HD, kc * 32, half), ca);
#pragma unroll
    for (int kc = 0; kc < 8; ++kc)
      cb = wmma_bf16(load_a_frag(&hA[m][0], kc * 32, half),
                     load_b_frag(ws + OFF_UWT, col, HD, kc * 32, half), cb);
    const float ub = U_b[col];
#pragma unroll
    for (int i = 0; i < 8; ++i)
      zrS[half * 8 + i][col] = 1.0f / (1.0f + __expf(-(ca[i] + cb[i] + ub)));
  }
  __syncthreads();

  // rgA = bf16(r * g)
  {
    const float* gp = g + (size_t)(bbase + prow) * HD + pk;
    float4 a0 = *(const float4*)(gp + 0);
    float4 a1 = *(const float4*)(gp + 4);
    float4 a2 = *(const float4*)(gp + 8);
    float4 a3 = *(const float4*)(gp + 12);
    const float* zp = &zrS[prow][HD + pk];
    float4 r0 = *(const float4*)(zp + 0);
    float4 r1 = *(const float4*)(zp + 4);
    float4 r2 = *(const float4*)(zp + 8);
    float4 r3 = *(const float4*)(zp + 12);
    float4 m0 = {r0.x * a0.x, r0.y * a0.y, r0.z * a0.z, r0.w * a0.w};
    float4 m1 = {r1.x * a1.x, r1.y * a1.y, r1.z * a1.z, r1.w * a1.w};
    float4 m2 = {r2.x * a2.x, r2.y * a2.y, r2.z * a2.z, r2.w * a2.w};
    float4 m3 = {r3.x * a3.x, r3.y * a3.y, r3.z * a3.z, r3.w * a3.w};
    *(v8bf*)&rgA[prow][pk]     = cvt_v8(m0, m1);
    *(v8bf*)&rgA[prow][pk + 8] = cvt_v8(m2, m3);
  }
  __syncthreads();

  // gt = tanh([r*g, h_avg] @ V_W + V_b); out = (1-z)g + z*gt
  for (int t = 0; t < 2; ++t) {
    const int col = wave * 32 + t * 16 + m;
    v8f ca = {}, cb = {};
#pragma unroll
    for (int kc = 0; kc < 8; ++kc)
      ca = wmma_bf16(load_a_frag(&rgA[m][0], kc * 32, half),
                     load_b_frag(ws + OFF_VWT, col, 2 * HD, kc * 32, half), ca);
#pragma unroll
    for (int kc = 0; kc < 8; ++kc)
      cb = wmma_bf16(load_a_frag(&hA[m][0], kc * 32, half),
                     load_b_frag(ws + OFF_VWT, col, 2 * HD, 256 + kc * 32, half),
                     cb);
    const float vb = V_b[col];
#pragma unroll
    for (int i = 0; i < 8; ++i) {
      const int row = half * 8 + i;
      const float z  = zrS[row][col];
      const float gv = g[(size_t)(bbase + row) * HD + col];
      const float gt = tanhf(ca[i] + cb[i] + vb);
      out[(size_t)(bbase + row) * HD + col] = (1.0f - z) * gv + z * gt;
    }
  }
}

// ---------------------------------------------------------------------------
extern "C" void kernel_launch(void* const* d_in, const int* in_sizes, int n_in,
                              void* d_out, int out_size, void* d_ws,
                              size_t ws_size, hipStream_t stream) {
  const float* g    = (const float*)d_in[0];
  const float* h    = (const float*)d_in[1];
  const int*   mask = (const int*)d_in[2];
  const float* w_W  = (const float*)d_in[3];
  const float* u_W  = (const float*)d_in[4];
  const float* u_b  = (const float*)d_in[5];
  const float* W_W  = (const float*)d_in[6];
  const float* U_W  = (const float*)d_in[7];
  const float* U_b  = (const float*)d_in[8];
  const float* V_W  = (const float*)d_in[9];
  const float* V_b  = (const float*)d_in[10];
  float*  out = (float*)d_out;
  bf16_t* ws  = (bf16_t*)d_ws;   // needs 1 MB (bf16 transposed weights)

  convert_weights<<<256, 256, 0, stream>>>(w_W, u_W, W_W, U_W, V_W, ws);
  attn_pool_kernel<<<dim3(BDIM / 16, 2), 256, 0, stream>>>(g, h, mask, u_b,
                                                           ws, out);
  gru_gate_kernel<<<BDIM / 16, 256, 0, stream>>>(g, U_b, V_b, ws, out);
}
